// GATSepNet_10797547782308
// MI455X (gfx1250) — matrix operand order repaired and checked by reading, not compiled, not run on hardware
//
#include <hip/hip_runtime.h>
#include <math.h>

typedef __attribute__((ext_vector_type(16))) __bf16 v16bf;
typedef __attribute__((ext_vector_type(8)))  float  v8f;

#define K_DIM 128
#define GAT_NEG_SLOPE 0.2f
#define ROWS_PER_BLK 32

// ---------------------------------------------------------------------------
// GEMM: Hout[N,M] = X[N,128] @ W[128,M]   (f32 in, bf16 WMMA, f32 accumulate)
// Block = 256 thr (8 waves) per 32-row strip. Both operands staged in LDS in
// WMMA *fragment order* so each fragment load is one contiguous 32B ds read.
// ---------------------------------------------------------------------------
template<int M>
__global__ __launch_bounds__(256) void gat_gemm_bf16(
    const float* __restrict__ X, const float* __restrict__ W,
    float* __restrict__ Hout)
{
    constexpr int NT  = M / 16;                 // col tiles (8 or 4)
    constexpr int MSH = (M == 128) ? 7 : 6;     // log2(M)
    // fragment-ordered LDS: [tile][kk][lane][e] of bf16
    __shared__ __attribute__((aligned(32))) __bf16 aFrag[2  * 4 * 32 * 16]; // 8 KB
    __shared__ __attribute__((aligned(32))) __bf16 bFrag[NT * 4 * 32 * 16]; // 32/16 KB

    const int rowBase = blockIdx.x * ROWS_PER_BLK;
    const int t = threadIdx.x;

    // ---- stage X (32x128 f32, coalesced) -> bf16 A fragments ----
    #pragma unroll
    for (int i = 0; i < (ROWS_PER_BLK * K_DIM) / 256; ++i) {    // 16 iters
        int linear = i * 256 + t;
        int r = linear >> 7;            // row in strip 0..31
        int c = linear & 127;           // k
        float v = X[(size_t)(rowBase + r) * K_DIM + c];
        // invert 16-bit A 16x32 layout: k = ((e&8)<<1) + (hi<<3) + (e&7)
        int rt = r >> 4, m = r & 15;
        int kk = c >> 5, kr = c & 31;
        int hi = (kr >> 3) & 1;
        int e  = ((kr >> 4) << 3) | (kr & 7);
        int lane = (hi << 4) | m;
        aFrag[(((rt * 4 + kk) * 32) + lane) * 16 + e] = (__bf16)v;
    }
    // ---- stage W (128xM f32, coalesced, L2-resident) -> bf16 B fragments ----
    #pragma unroll
    for (int i = 0; i < (K_DIM * M) / 256; ++i) {               // 64/32 iters
        int linear = i * 256 + t;
        int k = linear >> MSH;
        int n = linear & (M - 1);
        float v = W[linear];
        // invert 16-bit B 32x16 layout: k = ((e&8)<<1) + (hi<<3) + (e&7)
        int ct = n >> 4, nl = n & 15;
        int kk = k >> 5, kr = k & 31;
        int hi = (kr >> 3) & 1;
        int e  = ((kr >> 4) << 3) | (kr & 7);
        int lane = (hi << 4) | nl;
        bFrag[(((ct * 4 + kk) * 32) + lane) * 16 + e] = (__bf16)v;
    }
    __syncthreads();

    const int wave = t >> 5;
    const int lane = t & 31;
    const int mloc = lane & 15;
    const int hiL  = lane >> 4;

    for (int idx = wave; idx < 2 * NT; idx += 8) {
        const int rt = idx / NT;
        const int ct = idx % NT;
        v8f acc = {};
        #pragma unroll
        for (int kk = 0; kk < 4; ++kk) {        // K = 4 * 32
            v16bf a = *(const v16bf*)&aFrag[(((rt * 4 + kk) * 32) + lane) * 16];
            v16bf b = *(const v16bf*)&bFrag[(((ct * 4 + kk) * 32) + lane) * 16];
            acc = __builtin_amdgcn_wmma_f32_16x16x32_bf16(
                false, a, false, b, (short)0, acc, false, false);
        }
        // C/D layout: element v, lane l -> col n = l&15, row m = (l>>4)*8 + v.
        // Single base address; the 8 row-strided stores use constant byte
        // offsets (v*M*4 <= 3584) folded into the instruction offset field.
        float* p = Hout + (size_t)(rowBase + rt * 16 + hiL * 8) * M + ct * 16 + mloc;
        #pragma unroll
        for (int v = 0; v < 8; ++v)
            p[v * M] = acc[v];
    }
}

// ---------------------------------------------------------------------------
// Attention logits: als[n,h] = <h[n,h,:], a_src[h,:]> ; ald likewise
// ---------------------------------------------------------------------------
template<int H, int C>
__global__ void gat_logits(const float* __restrict__ Hf,
                           const float* __restrict__ a_src,
                           const float* __restrict__ a_dst,
                           float* __restrict__ als, float* __restrict__ ald,
                           int Nn)
{
    int tid = blockIdx.x * blockDim.x + threadIdx.x;
    if (tid >= Nn * H) return;
    int n = tid / H, hd = tid % H;
    const float* hr = Hf + (size_t)n * (H * C) + hd * C;
    float s = 0.f, d = 0.f;
    #pragma unroll 8
    for (int c = 0; c < C; ++c) {
        float v = hr[c];
        s = fmaf(v, a_src[hd * C + c], s);
        d = fmaf(v, a_dst[hd * C + c], d);
    }
    als[tid] = s;
    ald[tid] = d;
}

// monotonic float<->uint encoding so segment-max can use integer atomicMax
__device__ __forceinline__ unsigned fenc(float f) {
    unsigned u = __float_as_uint(f);
    return (u >> 31) ? ~u : (u | 0x80000000u);
}
__device__ __forceinline__ float fdec(unsigned u) {
    u = (u >> 31) ? (u ^ 0x80000000u) : ~u;
    return __uint_as_float(u);
}

// ---------------------------------------------------------------------------
// Pass 1 over edges: segment max of leaky_relu(als[src]+ald[dst]) into emax[dst]
// ---------------------------------------------------------------------------
template<int H>
__global__ void gat_edge_max(const int* __restrict__ esrc, const int* __restrict__ edst,
                             int Ee, int Nn,
                             const float* __restrict__ als, const float* __restrict__ ald,
                             unsigned* __restrict__ emax)
{
    int tid = blockIdx.x * blockDim.x + threadIdx.x;
    int Etot = Ee + Nn;
    if (tid >= Etot * H) return;
    int e = tid / H, hd = tid % H;
    int s, d;
    if (e < Ee) { s = esrc[e]; d = edst[e]; } else { s = d = e - Ee; }
    float v = als[s * H + hd] + ald[d * H + hd];
    v = v > 0.f ? v : v * GAT_NEG_SLOPE;
    atomicMax(&emax[d * H + hd], fenc(v));
}

// ---------------------------------------------------------------------------
// Pass 2 over edges: w = exp(e - emax[dst]); denom[dst] += w; acc[dst] += w*h[src]
// one wave32 per edge; lane owns D/32 contiguous channels (float4 / float2)
// ---------------------------------------------------------------------------
template<int H, int C>
__global__ __launch_bounds__(256) void gat_edge_accum(
    const int* __restrict__ esrc, const int* __restrict__ edst,
    int Ee, int Nn,
    const float* __restrict__ als, const float* __restrict__ ald,
    const unsigned* __restrict__ emax,
    const float* __restrict__ Hf,
    float* __restrict__ denom, float* __restrict__ acc)
{
    constexpr int D = H * C;
    constexpr int VEC = D / 32;
    int gw = (int)((blockIdx.x * blockDim.x + threadIdx.x) >> 5);   // edge id
    int lane = threadIdx.x & 31;
    int Etot = Ee + Nn;
    if (gw >= Etot) return;
    int s, d;
    if (gw < Ee) { s = esrc[gw]; d = edst[gw]; } else { s = d = gw - Ee; }

    const int c0 = lane * VEC;
    const int hd = c0 / C;
    float v = als[s * H + hd] + ald[d * H + hd];
    v = v > 0.f ? v : v * GAT_NEG_SLOPE;
    float m = fdec(emax[d * H + hd]);
    float w = __expf(v - m);

    if ((c0 & (C - 1)) == 0)                 // one lane per head
        atomicAdd(&denom[d * H + hd], w);

    const float* hsrc = Hf + (size_t)s * D + c0;
    float*       aout = acc + (size_t)d * D + c0;
    if constexpr (VEC == 4) {
        const float4 hv = *reinterpret_cast<const float4*>(hsrc);
        atomicAdd(aout + 0, w * hv.x);
        atomicAdd(aout + 1, w * hv.y);
        atomicAdd(aout + 2, w * hv.z);
        atomicAdd(aout + 3, w * hv.w);
    } else {
        const float2 hv = *reinterpret_cast<const float2*>(hsrc);
        atomicAdd(aout + 0, w * hv.x);
        atomicAdd(aout + 1, w * hv.y);
    }
}

// ---------------------------------------------------------------------------
// Finalize: out = acc/(denom+1e-16) + bias ; optional ELU
// ---------------------------------------------------------------------------
template<int H, int C, bool ELU>
__global__ void gat_finalize(const float* __restrict__ acc,
                             const float* __restrict__ denom,
                             const float* __restrict__ bias,
                             float* __restrict__ out, int Nn)
{
    constexpr int D = H * C;
    int tid = blockIdx.x * blockDim.x + threadIdx.x;
    if (tid >= Nn * D) return;
    int n = tid / D, c = tid % D;
    int hd = c / C;
    float v = acc[tid] / (denom[n * H + hd] + 1e-16f) + bias[c];
    if (ELU) v = v > 0.f ? v : (__expf(v) - 1.f);
    out[tid] = v;
}

// ---------------------------------------------------------------------------
extern "C" void kernel_launch(void* const* d_in, const int* in_sizes, int n_in,
                              void* d_out, int out_size, void* d_ws, size_t ws_size,
                              hipStream_t stream)
{
    (void)in_sizes; (void)n_in; (void)out_size; (void)ws_size;

    const int N = 100000, E = 1000000;
    const int Etot = E + N;

    const float* x   = (const float*)d_in[0];
    const int*   ei  = (const int*)  d_in[1];
    const float* W0  = (const float*)d_in[2];
    const float* as0 = (const float*)d_in[3];
    const float* ad0 = (const float*)d_in[4];
    const float* b0  = (const float*)d_in[5];
    const float* W1  = (const float*)d_in[6];
    const float* as1 = (const float*)d_in[7];
    const float* ad1 = (const float*)d_in[8];
    const float* b1  = (const float*)d_in[9];
    const float* W2  = (const float*)d_in[10];
    const float* as2 = (const float*)d_in[11];
    const float* ad2 = (const float*)d_in[12];
    const float* b2  = (const float*)d_in[13];

    const int* esrc = ei;
    const int* edst = ei + E;

    // workspace layout (f32 unless noted)
    float*    hbuf  = (float*)d_ws;                       // N*128
    float*    accb  = hbuf + (size_t)N * 128;             // N*128
    float*    xb    = accb + (size_t)N * 128;             // N*128
    float*    als   = xb   + (size_t)N * 128;             // N*4
    float*    ald   = als  + (size_t)N * 4;               // N*4
    unsigned* emax  = (unsigned*)(ald + (size_t)N * 4);   // N*4
    float*    denom = (float*)(emax + (size_t)N * 4);     // N*4

    const int BLK = 256;
    const int gemmGrid = N / ROWS_PER_BLK;                // 100000 % 32 == 0
    const int accGrid  = (Etot + 7) / 8;                  // 8 waves / block

    // ---------------- layer 0: 128 -> 4 heads x 32, concat, ELU -------------
    gat_gemm_bf16<128><<<gemmGrid, BLK, 0, stream>>>(x, W0, hbuf);
    gat_logits<4, 32><<<(N * 4 + BLK - 1) / BLK, BLK, 0, stream>>>(hbuf, as0, ad0, als, ald, N);
    hipMemsetAsync(emax,  0, (size_t)N * 4 * sizeof(unsigned), stream);
    hipMemsetAsync(denom, 0, (size_t)N * 4 * sizeof(float),    stream);
    hipMemsetAsync(accb,  0, (size_t)N * 128 * sizeof(float),  stream);
    gat_edge_max<4><<<(Etot * 4 + BLK - 1) / BLK, BLK, 0, stream>>>(esrc, edst, E, N, als, ald, emax);
    gat_edge_accum<4, 32><<<accGrid, BLK, 0, stream>>>(esrc, edst, E, N, als, ald, emax, hbuf, denom, accb);
    gat_finalize<4, 32, true><<<(N * 128 + BLK - 1) / BLK, BLK, 0, stream>>>(accb, denom, b0, xb, N);

    // ---------------- layer 1: 128 -> 4 heads x 32, concat, ELU -------------
    gat_gemm_bf16<128><<<gemmGrid, BLK, 0, stream>>>(xb, W1, hbuf);
    gat_logits<4, 32><<<(N * 4 + BLK - 1) / BLK, BLK, 0, stream>>>(hbuf, as1, ad1, als, ald, N);
    hipMemsetAsync(emax,  0, (size_t)N * 4 * sizeof(unsigned), stream);
    hipMemsetAsync(denom, 0, (size_t)N * 4 * sizeof(float),    stream);
    hipMemsetAsync(accb,  0, (size_t)N * 128 * sizeof(float),  stream);
    gat_edge_max<4><<<(Etot * 4 + BLK - 1) / BLK, BLK, 0, stream>>>(esrc, edst, E, N, als, ald, emax);
    gat_edge_accum<4, 32><<<accGrid, BLK, 0, stream>>>(esrc, edst, E, N, als, ald, emax, hbuf, denom, accb);
    gat_finalize<4, 32, true><<<(N * 128 + BLK - 1) / BLK, BLK, 0, stream>>>(accb, denom, b1, xb, N);

    // ---------------- layer 2: 128 -> 1 head x 64, no ELU, -> d_out ---------
    gat_gemm_bf16<64><<<gemmGrid, BLK, 0, stream>>>(xb, W2, hbuf);
    gat_logits<1, 64><<<(N + BLK - 1) / BLK, BLK, 0, stream>>>(hbuf, as2, ad2, als, ald, N);
    hipMemsetAsync(emax,  0, (size_t)N * sizeof(unsigned), stream);
    hipMemsetAsync(denom, 0, (size_t)N * sizeof(float),    stream);
    hipMemsetAsync(accb,  0, (size_t)N * 64 * sizeof(float), stream);
    gat_edge_max<1><<<(Etot + BLK - 1) / BLK, BLK, 0, stream>>>(esrc, edst, E, N, als, ald, emax);
    gat_edge_accum<1, 64><<<accGrid, BLK, 0, stream>>>(esrc, edst, E, N, als, ald, emax, hbuf, denom, accb);
    gat_finalize<1, 64, false><<<(N * 64 + BLK - 1) / BLK, BLK, 0, stream>>>(accb, denom, b2, (float*)d_out, N);
}
